// MultiHeadAttention_21139829031074
// MI455X (gfx1250) — matrix operand order repaired and checked by reading, not compiled, not run on hardware
//
#include <hip/hip_runtime.h>
#include <hip/hip_bf16.h>

typedef _Float16 h16;
typedef __attribute__((ext_vector_type(16))) _Float16 v16h;
typedef __attribute__((ext_vector_type(8)))  _Float16 v8h;
typedef __attribute__((ext_vector_type(8)))  float    v8f;

#define D_MODEL 1024
#define BATCH   2
#define SEQ     2048
#define NH      16
#define DH      64
#define NTOK    (BATCH * SEQ)   // 4096

// ---------------------------------------------------------------------------
// Fragment loaders (CDNA5 WMMA 16x16x32 f16 layouts, wave32)
//
// A (16x32, MxK): lane L (m = L&15, hi = L>>4) holds row m.
//   v16h element i -> K = i + 8*hi (i<8) ; K = 8 + i + 8*hi (i>=8)
// B (32x16, KxN): lane L (n = L&15, hi = L>>4) holds column n.
//   v16h element i -> K = i + 8*hi (i<8) ; K = 8 + i + 8*hi (i>=8)   (B stored transposed: N x K row-major)
// C/D (16x16 f32): lane (n = L&15, hi = L>>4), VGPR r -> row m = r + 8*hi.
// ---------------------------------------------------------------------------

__device__ __forceinline__ v16h load_a16x32(const h16* __restrict__ tile, int ld) {
    const int lane = threadIdx.x & 31;
    const int m  = lane & 15;
    const int hi = lane >> 4;
    const h16* p = tile + m * ld + 8 * hi;
    union { v16h v; v8h h[2]; } u;
    u.h[0] = *(const v8h*)(p);
    u.h[1] = *(const v8h*)(p + 16);
    return u.v;
}

// btile points at BT[n0][k0] of an N x K row-major (transposed-B) array
__device__ __forceinline__ v16h load_b32x16(const h16* __restrict__ btile, int ld) {
    const int lane = threadIdx.x & 31;
    const int n  = lane & 15;
    const int hi = lane >> 4;
    const h16* p = btile + n * ld + 8 * hi;
    union { v16h v; v8h h[2]; } u;
    u.h[0] = *(const v8h*)(p);
    u.h[1] = *(const v8h*)(p + 16);
    return u.v;
}

// ---------------------------------------------------------------------------
// Prep kernels
// ---------------------------------------------------------------------------
__global__ void cvt_f16_kernel(const float* __restrict__ src, h16* __restrict__ dst, int n) {
    int i = blockIdx.x * blockDim.x + threadIdx.x;
    if (i < n) dst[i] = (h16)src[i];
}

// src: [K=D][N=D] f32 row-major ; dst: [N=D][K=D] f16 row-major (transposed)
__global__ void transpose_cvt_kernel(const float* __restrict__ src, h16* __restrict__ dst) {
    int i = blockIdx.x * blockDim.x + threadIdx.x;
    if (i < D_MODEL * D_MODEL) {
        int n = i / D_MODEL;
        int k = i % D_MODEL;
        dst[i] = (h16)src[k * D_MODEL + n];
    }
}

// ---------------------------------------------------------------------------
// WMMA GEMM: out[M x D] = A[M x D] @ B[D x D], B given transposed (BT: N x K).
//
// Block = 256 threads (8 waves) -> 128 x 64 output tile.
//   Wave w computes rows [m0+16w, m0+16w+16) x all 64 block columns
//   (4 accumulators, one A-fragment shared across 4 WMMAs per k-step).
//   The 64x32 B tile per k-step is DMA'd into double-buffered LDS with
//   global_load_async_to_lds_b128 (ASYNCcnt), prefetching the next buffer
//   while the current one feeds the matrix pipe. Loop unrolled x2 so the
//   buffer parity is compile-time constant.
//
// MODE: 0 = f16 row-major out, 1 = f16 per-head transposed out
//       (VT[(b*NH+h)*DH + dh][s]), 2 = f32 out + bias.
// ---------------------------------------------------------------------------
template <int MODE>
__global__ __launch_bounds__(256) void gemm_wmma_kernel(
        const h16* __restrict__ A, const h16* __restrict__ BT,
        void* __restrict__ out, const float* __restrict__ bias, int M) {
    __shared__ h16 bstage[2][64 * 32];      // double-buffered B tile, 2 x 4 KB

    const int tid  = threadIdx.x;
    const int wave = tid >> 5;
    const int lane = tid & 31;

    const int nblkN = D_MODEL / 64;          // 16 column blocks
    const int mb = blockIdx.x / nblkN;
    const int nbk = blockIdx.x % nblkN;
    const int m0 = mb * 128 + wave * 16;     // this wave's 16 rows
    const int n0 = nbk * 64;                 // block's 64 columns

    const h16* arow = A + (size_t)m0 * D_MODEL;

    // B staging: thread t copies 16B of BT[n0 + t/4][(t%4)*8 + k0]
    const int brow = tid >> 2;               // 0..63
    const int bcol = (tid & 3) * 8;          // 0,8,16,24
    const h16* bsrc = BT + (size_t)(n0 + brow) * D_MODEL + bcol;
    // low 32 bits of a generic LDS pointer == LDS byte offset on gfx1250
    const unsigned ldsOff0 = (unsigned)(uintptr_t)&bstage[0][brow * 32 + bcol];
    const unsigned ldsOff1 = (unsigned)(uintptr_t)&bstage[1][brow * 32 + bcol];

    // prefetch k0 = 0 into buffer 0
    asm volatile("global_load_async_to_lds_b128 %0, %1, off"
                 :: "v"(ldsOff0),
                    "v"((unsigned long long)(uintptr_t)bsrc)
                 : "memory");

    v8f acc0 = {}, acc1 = {}, acc2 = {}, acc3 = {};

#pragma unroll 2
    for (int k0 = 0; k0 < D_MODEL; k0 += 32) {
        const int cur = (k0 >> 5) & 1;

        // my async DMA done, then block-wide: everyone's DMA done
        asm volatile("s_wait_asynccnt 0" ::: "memory");
        __syncthreads();

        // kick off DMA of the next k-step into the other buffer
        if (k0 + 32 < D_MODEL) {
            const unsigned dstOff = cur ? ldsOff0 : ldsOff1;
            asm volatile("global_load_async_to_lds_b128 %0, %1, off"
                         :: "v"(dstOff),
                            "v"((unsigned long long)(uintptr_t)(bsrc + k0 + 32))
                         : "memory");
        }

        v16h a = load_a16x32(arow + k0, D_MODEL);
        const h16* bl = &bstage[cur][0];
        v16h b0 = load_b32x16(bl + 0 * 16 * 32, 32);
        v16h b1 = load_b32x16(bl + 1 * 16 * 32, 32);
        v16h b2 = load_b32x16(bl + 2 * 16 * 32, 32);
        v16h b3 = load_b32x16(bl + 3 * 16 * 32, 32);

        acc0 = __builtin_amdgcn_wmma_f32_16x16x32_f16(false, a, false, b0, (short)0, acc0, false, false);
        acc1 = __builtin_amdgcn_wmma_f32_16x16x32_f16(false, a, false, b1, (short)0, acc1, false, false);
        acc2 = __builtin_amdgcn_wmma_f32_16x16x32_f16(false, a, false, b2, (short)0, acc2, false, false);
        acc3 = __builtin_amdgcn_wmma_f32_16x16x32_f16(false, a, false, b3, (short)0, acc3, false, false);
    }

    const int n  = lane & 15;
    const int hi = lane >> 4;
    v8f accs[4] = {acc0, acc1, acc2, acc3};

#pragma unroll
    for (int nt = 0; nt < 4; ++nt) {
        const int nc = n0 + nt * 16;
        if (MODE == 0) {
            h16* o = (h16*)out;
#pragma unroll
            for (int r = 0; r < 8; ++r)
                o[(size_t)(m0 + r + 8 * hi) * D_MODEL + nc + n] = (h16)accs[nt][r];
        } else if (MODE == 1) {
            // per-head transposed V: VT[((b*NH + h)*DH + dh)][s], s within batch
            h16* o = (h16*)out;
            const int b   = m0 / SEQ;        // 16-row tile never straddles a batch
            const int h   = nc / DH;         // 16-col tile never straddles a head
            const int dh0 = nc % DH;
#pragma unroll
            for (int r = 0; r < 8; ++r) {
                int s = (m0 % SEQ) + r + 8 * hi;
                o[((size_t)(b * NH + h) * DH + dh0 + n) * SEQ + s] = (h16)accs[nt][r];
            }
        } else {
            float* o = (float*)out;
            const float bb = bias[nc + n];
#pragma unroll
            for (int r = 0; r < 8; ++r)
                o[(size_t)(m0 + r + 8 * hi) * D_MODEL + nc + n] = accs[nt][r] + bb;
        }
    }
}

// ---------------------------------------------------------------------------
// Flash attention: one wave per (batch, head, 16-query tile).
// Q, K: f16 row-major [tok][1024] (head slab at col h*64). Q is pre-scaled
// by 1/sqrt(DH) at fragment load. Exactly one key block (the last) needs
// the causal mask; all earlier blocks run a mask-free fast path.
// VT:   f16 [(b*NH+h)*DH + dh][SEQ].
// CTX:  f16 row-major [tok][1024].
// ---------------------------------------------------------------------------
__global__ __launch_bounds__(256) void attn_wmma_kernel(
        const h16* __restrict__ Q, const h16* __restrict__ K,
        const h16* __restrict__ VT, h16* __restrict__ CTX) {
    __shared__ h16 plds[8][16 * 32];      // per-wave 16x32 P staging tile (8 KB)

    const int widb = threadIdx.x >> 5;    // wave within block
    const int wid  = (blockIdx.x * blockDim.x + threadIdx.x) >> 5;
    const int qt = wid % (SEQ / 16);
    const int bh = wid / (SEQ / 16);      // b*NH + h
    const int b  = bh / NH;
    const int h  = bh % NH;
    const int q0 = qt * 16;

    const int lane = threadIdx.x & 31;
    const int nn = lane & 15;
    const int hi = lane >> 4;

    const h16* qbase = Q + ((size_t)(b * SEQ + q0)) * D_MODEL + h * DH;
    const h16* kbase = K + ((size_t)(b * SEQ)) * D_MODEL + h * DH;
    const h16* vbase = VT + ((size_t)bh * DH) * SEQ;

    // Q A-fragments (16 queries x 64 dh), loaded once, pre-scaled by 1/8
    const v16h aq0 = load_a16x32(qbase, D_MODEL) * (h16)0.125f;
    const v16h aq1 = load_a16x32(qbase + 32, D_MODEL) * (h16)0.125f;

    float rm[8], rl[8];
#pragma unroll
    for (int r = 0; r < 8; ++r) { rm[r] = -3.0e30f; rl[r] = 0.0f; }
    v8f o0 = {}, o1 = {}, o2 = {}, o3 = {};

    h16* myp = &plds[widb][0];

    auto process_block = [&](int kp0, bool domask) {
        // scores: two 16x16 tiles over keys [kp0, kp0+32)
        v8f sl = {}, sh = {};
        sl = __builtin_amdgcn_wmma_f32_16x16x32_f16(false, aq0, false,
                load_b32x16(kbase + (size_t)kp0 * D_MODEL, D_MODEL),
                (short)0, sl, false, false);
        sl = __builtin_amdgcn_wmma_f32_16x16x32_f16(false, aq1, false,
                load_b32x16(kbase + (size_t)kp0 * D_MODEL + 32, D_MODEL),
                (short)0, sl, false, false);
        sh = __builtin_amdgcn_wmma_f32_16x16x32_f16(false, aq0, false,
                load_b32x16(kbase + (size_t)(kp0 + 16) * D_MODEL, D_MODEL),
                (short)0, sh, false, false);
        sh = __builtin_amdgcn_wmma_f32_16x16x32_f16(false, aq1, false,
                load_b32x16(kbase + (size_t)(kp0 + 16) * D_MODEL + 32, D_MODEL),
                (short)0, sh, false, false);

        // causal mask (tail block only) + online softmax
        float mnew[8];
#pragma unroll
        for (int r = 0; r < 8; ++r) {
            float x0 = sl[r];
            float x1 = sh[r];
            if (domask) {
                const int q = q0 + r + 8 * hi;
                if (kp0 + nn      > q) x0 = -3.0e30f;
                if (kp0 + 16 + nn > q) x1 = -3.0e30f;
                sl[r] = x0; sh[r] = x1;
            }
            float t = fmaxf(x0, x1);
#pragma unroll
            for (int off = 1; off < 16; off <<= 1)
                t = fmaxf(t, __shfl_xor(t, off, 32));
            mnew[r] = fmaxf(rm[r], t);
        }

#pragma unroll
        for (int r = 0; r < 8; ++r) {
            const float sc = __expf(rm[r] - mnew[r]);
            const float pl = __expf(sl[r] - mnew[r]);
            const float ph = __expf(sh[r] - mnew[r]);
            float rs = pl + ph;
#pragma unroll
            for (int off = 1; off < 16; off <<= 1)
                rs += __shfl_xor(rs, off, 32);
            rl[r] = rl[r] * sc + rs;
            rm[r] = mnew[r];
            o0[r] *= sc; o1[r] *= sc; o2[r] *= sc; o3[r] *= sc;
            const int m = r + 8 * hi;
            myp[m * 32 + nn]      = (h16)pl;
            myp[m * 32 + 16 + nn] = (h16)ph;
        }

        // same-wave DS ordering: make P stores visible before transposed read
        asm volatile("s_wait_dscnt 0" ::: "memory");

        // re-read P in A-fragment layout (16 x 32 over keys)
        v16h ap;
        {
            const int m = lane & 15;
            const h16* p = myp + m * 32 + 8 * hi;
            union { v16h v; v8h h2[2]; } u;
            u.h2[0] = *(const v8h*)(p);
            u.h2[1] = *(const v8h*)(p + 16);
            ap = u.v;
        }

        // ctx += P @ V  (4 N-tiles of 16 over dh)
        o0 = __builtin_amdgcn_wmma_f32_16x16x32_f16(false, ap, false,
                load_b32x16(vbase + (size_t)0  * SEQ + kp0, SEQ), (short)0, o0, false, false);
        o1 = __builtin_amdgcn_wmma_f32_16x16x32_f16(false, ap, false,
                load_b32x16(vbase + (size_t)16 * SEQ + kp0, SEQ), (short)0, o1, false, false);
        o2 = __builtin_amdgcn_wmma_f32_16x16x32_f16(false, ap, false,
                load_b32x16(vbase + (size_t)32 * SEQ + kp0, SEQ), (short)0, o2, false, false);
        o3 = __builtin_amdgcn_wmma_f32_16x16x32_f16(false, ap, false,
                load_b32x16(vbase + (size_t)48 * SEQ + kp0, SEQ), (short)0, o3, false, false);
    };

    // unmasked full blocks, then the single partially-masked tail block
    const int fullb = q0 >> 5;
    for (int kb = 0; kb < fullb; ++kb)
        process_block(kb * 32, false);
    process_block(fullb * 32, true);

    // normalize and store ctx (f16 row-major [tok][1024])
#pragma unroll
    for (int r = 0; r < 8; ++r) {
        const float inv = 1.0f / rl[r];
        const int tok = b * SEQ + q0 + r + 8 * hi;
        h16* orow = CTX + (size_t)tok * D_MODEL + h * DH;
        orow[0  + nn] = (h16)(o0[r] * inv);
        orow[16 + nn] = (h16)(o1[r] * inv);
        orow[32 + nn] = (h16)(o2[r] * inv);
        orow[48 + nn] = (h16)(o3[r] * inv);
    }
}

// ---------------------------------------------------------------------------
extern "C" void kernel_launch(void* const* d_in, const int* in_sizes, int n_in,
                              void* d_out, int out_size, void* d_ws, size_t ws_size,
                              hipStream_t stream) {
    const float* x  = (const float*)d_in[0];
    const float* wq = (const float*)d_in[1];
    const float* wk = (const float*)d_in[2];
    const float* wv = (const float*)d_in[3];
    const float* wo = (const float*)d_in[4];
    const float* bo = (const float*)d_in[5];
    float* out = (float*)d_out;

    h16* ws = (h16*)d_ws;
    h16* Xh  = ws; ws += (size_t)NTOK * D_MODEL;
    h16* WqT = ws; ws += (size_t)D_MODEL * D_MODEL;
    h16* WkT = ws; ws += (size_t)D_MODEL * D_MODEL;
    h16* WvT = ws; ws += (size_t)D_MODEL * D_MODEL;
    h16* WoT = ws; ws += (size_t)D_MODEL * D_MODEL;
    h16* Qh  = ws; ws += (size_t)NTOK * D_MODEL;
    h16* Kh  = ws; ws += (size_t)NTOK * D_MODEL;
    h16* VT  = ws; ws += (size_t)NTOK * D_MODEL;
    h16* CTX = ws; ws += (size_t)NTOK * D_MODEL;

    // 1. precision / layout prep
    cvt_f16_kernel<<<(NTOK * D_MODEL) / 256, 256, 0, stream>>>(x, Xh, NTOK * D_MODEL);
    transpose_cvt_kernel<<<(D_MODEL * D_MODEL) / 256, 256, 0, stream>>>(wq, WqT);
    transpose_cvt_kernel<<<(D_MODEL * D_MODEL) / 256, 256, 0, stream>>>(wk, WkT);
    transpose_cvt_kernel<<<(D_MODEL * D_MODEL) / 256, 256, 0, stream>>>(wv, WvT);
    transpose_cvt_kernel<<<(D_MODEL * D_MODEL) / 256, 256, 0, stream>>>(wo, WoT);

    // 2. QKV projections: 512 blocks of 128x64 tiles, async-LDS B staging
    const int gemm_blocks = (NTOK / 128) * (D_MODEL / 64);   // 512
    gemm_wmma_kernel<0><<<gemm_blocks, 256, 0, stream>>>(Xh, WqT, (void*)Qh, nullptr, NTOK);
    gemm_wmma_kernel<0><<<gemm_blocks, 256, 0, stream>>>(Xh, WkT, (void*)Kh, nullptr, NTOK);
    gemm_wmma_kernel<1><<<gemm_blocks, 256, 0, stream>>>(Xh, WvT, (void*)VT, nullptr, NTOK);

    // 3. causal flash attention
    const int attn_blocks = BATCH * NH * (SEQ / 16) * 32 / 256;        // 512
    attn_wmma_kernel<<<attn_blocks, 256, 0, stream>>>(Qh, Kh, VT, CTX);

    // 4. output projection + bias (f32 out)
    gemm_wmma_kernel<2><<<gemm_blocks, 256, 0, stream>>>(CTX, WoT, (void*)out, bo, NTOK);
}
